// BitNetLayer_4887672783063
// MI455X (gfx1250) — matrix operand order repaired and checked by reading, not compiled, not run on hardware
//
#include <hip/hip_runtime.h>
#include <hip/hip_bf16.h>

// ---------------------------------------------------------------------------
// BitNet linear: out = (in @ sign(W)^T) * rowscale * scale_param
//   M = 8192 (B*S), N = 4096 (D_OUT), K = 4096 (D_IN)
// Memory-bound (~330MB @ 23.3 TB/s ~ 14us); f16 operands fit in 192MB L2.
// Path: f16 WMMA (v_wmma_f32_16x16x32_f16) + async global->LDS double buffer.
// ---------------------------------------------------------------------------

typedef _Float16 v16h __attribute__((ext_vector_type(16)));
typedef _Float16 h4   __attribute__((ext_vector_type(4)));
typedef float    v8f  __attribute__((ext_vector_type(8)));
typedef int      v4i  __attribute__((ext_vector_type(4)));

union FragH { v16h h; uint4 u[2]; };

#define M_DIM 8192
#define N_DIM 4096
#define K_DIM 4096

#define BM 128
#define BN 128
#define BK 64
#define LDST 72   // padded LDS row stride in halves (144 B, 16B-aligned rows)

// --------------------------- async copy plumbing ---------------------------
#if __has_builtin(__builtin_amdgcn_global_load_async_to_lds_b128)
#define HAS_ASYNC_LDS 1
#else
#define HAS_ASYNC_LDS 0
#warning "gfx1250 async-to-LDS builtin not available; using sync LDS staging"
#endif

#if HAS_ASYNC_LDS
typedef __attribute__((address_space(1))) v4i* gv4i_ptr;  // global int4*
typedef __attribute__((address_space(3))) v4i* lv4i_ptr;  // LDS int4*
#endif

__device__ __forceinline__ void copy16_to_lds(const _Float16* g, _Float16* l) {
#if HAS_ASYNC_LDS
  // GLOBAL_LOAD_ASYNC_TO_LDS_B128: per-lane 16B memory->LDS, ASYNCcnt-tracked.
  __builtin_amdgcn_global_load_async_to_lds_b128(
      (gv4i_ptr)g, (lv4i_ptr)l, /*imm offset*/ 0, /*cpol*/ 0);
#else
  *(uint4*)l = *(const uint4*)g;
#endif
}

__device__ __forceinline__ void wait_async_copies() {
#if HAS_ASYNC_LDS
#if __has_builtin(__builtin_amdgcn_s_wait_asynccnt)
  __builtin_amdgcn_s_wait_asynccnt(0);
#else
  asm volatile("s_wait_asynccnt 0x0" ::: "memory");
#endif
#endif
}

// ---------------------------------------------------------------------------
// Phase 1: per-row abs-mean scale + binarize weights to f16 {-1,0,+1},
// keeping [o][k] row-major layout (exactly the B-fragment-friendly layout).
// ---------------------------------------------------------------------------
__global__ __launch_bounds__(256) void bitnet_binarize_kernel(
    const float* __restrict__ W, const float* __restrict__ scale_param,
    _Float16* __restrict__ wb, float* __restrict__ cscale) {
  __shared__ float red[256];
  const int o = blockIdx.x;
  const float* __restrict__ row = W + (size_t)o * K_DIM;
  _Float16* __restrict__ brow = wb + (size_t)o * K_DIM;

  float sum = 0.0f;
  for (int j = threadIdx.x * 4; j < K_DIM; j += 256 * 4) {
    float4 w = *(const float4*)(row + j);
    sum += fabsf(w.x) + fabsf(w.y) + fabsf(w.z) + fabsf(w.w);
    h4 b;
    b.x = (_Float16)((w.x > 0.0f) ? 1.0f : (w.x < 0.0f ? -1.0f : 0.0f));
    b.y = (_Float16)((w.y > 0.0f) ? 1.0f : (w.y < 0.0f ? -1.0f : 0.0f));
    b.z = (_Float16)((w.z > 0.0f) ? 1.0f : (w.z < 0.0f ? -1.0f : 0.0f));
    b.w = (_Float16)((w.w > 0.0f) ? 1.0f : (w.w < 0.0f ? -1.0f : 0.0f));
    *(h4*)(brow + j) = b;
  }
  red[threadIdx.x] = sum;
  __syncthreads();
  for (int s = 128; s > 0; s >>= 1) {
    if (threadIdx.x < (unsigned)s) red[threadIdx.x] += red[threadIdx.x + s];
    __syncthreads();
  }
  if (threadIdx.x == 0) {
    float rs = red[0] * (1.0f / (float)K_DIM);
    rs = fmaxf(rs, 1e-8f);
    cscale[o] = rs * scale_param[o];
  }
}

// ---------------------------------------------------------------------------
// Phase 2: fp32 -> f16 input conversion (row-major [m][k], A-fragment layout).
// ---------------------------------------------------------------------------
__global__ __launch_bounds__(256) void bitnet_cvt_kernel(
    const float* __restrict__ x, _Float16* __restrict__ y) {
  const size_t i = ((size_t)blockIdx.x * 256 + threadIdx.x) * 4;
  float4 v = *(const float4*)(x + i);
  h4 o;
  o.x = (_Float16)v.x; o.y = (_Float16)v.y;
  o.z = (_Float16)v.z; o.w = (_Float16)v.w;
  *(h4*)(y + i) = o;
}

// ---------------------------------------------------------------------------
// Phase 3: WMMA GEMM.  Block tile 128x128, BK=64, double-buffered LDS with
// async global->LDS copies.  8 wave32 in 2(M)x4(N); each wave: 4x2 tiles of
// 16x16 -> 8 v8f accumulators, 16 v_wmma_f32_16x16x32_f16 per K-stage.
// ---------------------------------------------------------------------------
__global__ __launch_bounds__(256) void bitnet_gemm_kernel(
    const _Float16* __restrict__ A,    // [M][K] f16 input
    const _Float16* __restrict__ Wb,   // [N][K] f16 binarized weight
    const float* __restrict__ cscale,  // [N]
    float* __restrict__ Out) {         // [M][N] f32
  __shared__ _Float16 As[2][BM * LDST];
  __shared__ _Float16 Bs[2][BN * LDST];

  const int bn = blockIdx.x * BN;
  const int bm = blockIdx.y * BM;
  const int tid = threadIdx.x;
  const int lane = tid & 31;
  const int wave = tid >> 5;
  const int wm = wave & 1;    // 0..1 : 64-row strip in M
  const int wn = wave >> 1;   // 0..3 : 32-col strip in N
  const int lrow = lane & 15;
  const int lhalf = lane >> 4;

  v8f acc[4][2];
#pragma unroll
  for (int mt = 0; mt < 4; ++mt)
#pragma unroll
    for (int nt = 0; nt < 2; ++nt)
      acc[mt][nt] = (v8f)(0.0f);

  // Tile loader: 256 threads * 16B = 32 rows/pass, 4 passes per 128-row tile.
  const int lr = tid >> 3;        // 0..31 (row within pass)
  const int lc = (tid & 7) * 8;   // 0..56 (half offset, 16B granules)

  const _Float16* aGlob = A  + (size_t)(bm + lr) * K_DIM + lc;
  const _Float16* bGlob = Wb + (size_t)(bn + lr) * K_DIM + lc;
  const int ldsOff = lr * LDST + lc;

  // Stage tile kt into buffer `buf` (async if available).
  auto stage = [&](int buf, int kt) {
#pragma unroll
    for (int p = 0; p < 4; ++p) {
      copy16_to_lds(aGlob + (size_t)p * 32 * K_DIM + kt,
                    &As[buf][ldsOff + p * 32 * LDST]);
      copy16_to_lds(bGlob + (size_t)p * 32 * K_DIM + kt,
                    &Bs[buf][ldsOff + p * 32 * LDST]);
    }
  };

  stage(0, 0);
  wait_async_copies();
  __syncthreads();

  int cur = 0;
  for (int kt = 0; kt < K_DIM; kt += BK) {
    const int nxt = cur ^ 1;
    if (kt + BK < K_DIM) stage(nxt, kt + BK);   // overlap copy with compute

#pragma unroll
    for (int ks = 0; ks < 2; ++ks) {
      const int kb = ks * 32;
      FragH af[4], bfr[2];
      // A fragment (16x32 f16, ISA 7.12.2): lane holds row M=lrow,
      // K chunks [8*lhalf .. +7] and [16+8*lhalf .. +7].
#pragma unroll
      for (int mt = 0; mt < 4; ++mt) {
        const _Float16* base =
            &As[cur][(wm * 64 + mt * 16 + lrow) * LDST + kb + 8 * lhalf];
        af[mt].u[0] = *(const uint4*)base;
        af[mt].u[1] = *(const uint4*)(base + 16);
      }
      // B fragment (32x16 f16): lane holds column N=lrow,
      // K chunks [8*lhalf .. +7] and [16+8*lhalf .. +7] (same order as A).
#pragma unroll
      for (int nt = 0; nt < 2; ++nt) {
        const _Float16* base =
            &Bs[cur][(wn * 32 + nt * 16 + lrow) * LDST + kb + 8 * lhalf];
        bfr[nt].u[0] = *(const uint4*)base;
        bfr[nt].u[1] = *(const uint4*)(base + 16);
      }
#pragma unroll
      for (int mt = 0; mt < 4; ++mt)
#pragma unroll
        for (int nt = 0; nt < 2; ++nt)
          acc[mt][nt] = __builtin_amdgcn_wmma_f32_16x16x32_f16(
              false, af[mt].h, false, bfr[nt].h,
              (short)0, acc[mt][nt], false, false);
    }

    wait_async_copies();   // this wave's copies into `nxt` done
    __syncthreads();       // all waves done reading `cur` / writing `nxt`
    cur = nxt;
  }

  // Epilogue: D layout (32-bit 16x16): lane = {N=lrow | +16: same N, M+8},
  // VGPR v -> row M = v + 8*lhalf.  Fused scale + nan_to_num.
#pragma unroll
  for (int nt = 0; nt < 2; ++nt) {
    const int n = bn + wn * 32 + nt * 16 + lrow;
    const float s = cscale[n];
#pragma unroll
    for (int mt = 0; mt < 4; ++mt) {
      const int rbase = bm + wm * 64 + mt * 16 + 8 * lhalf;
#pragma unroll
      for (int v = 0; v < 8; ++v) {
        float val = acc[mt][nt][v] * s;
        if (val != val) val = 0.0f;                       // NaN -> 0
        else if (val ==  __builtin_inff()) val =  1e6f;   // +inf -> 1e6
        else if (val == -__builtin_inff()) val = -1e6f;   // -inf -> -1e6
        Out[(size_t)(rbase + v) * N_DIM + n] = val;
      }
    }
  }
}

// ---------------------------------------------------------------------------
extern "C" void kernel_launch(void* const* d_in, const int* in_sizes, int n_in,
                              void* d_out, int out_size, void* d_ws, size_t ws_size,
                              hipStream_t stream) {
  (void)in_sizes; (void)n_in; (void)out_size; (void)ws_size;
  const float* input       = (const float*)d_in[0];  // [8192][4096]
  const float* weight      = (const float*)d_in[1];  // [4096][4096]
  const float* scale_param = (const float*)d_in[2];  // [4096]
  float* out = (float*)d_out;                        // [8192][4096]

  // Workspace layout: wb (32MB f16) | ain (64MB f16) | cscale (16KB f32)
  _Float16* wb  = (_Float16*)d_ws;
  _Float16* ain = (_Float16*)((char*)d_ws + (size_t)N_DIM * K_DIM * sizeof(_Float16));
  float* cscale = (float*)((char*)d_ws + (size_t)N_DIM * K_DIM * sizeof(_Float16)
                                       + (size_t)M_DIM * K_DIM * sizeof(_Float16));

  // Phase 1: row scales + binarized f16 weights.
  bitnet_binarize_kernel<<<N_DIM, 256, 0, stream>>>(weight, scale_param, wb, cscale);

  // Phase 2: input fp32 -> f16.
  const size_t nIn = (size_t)M_DIM * K_DIM;
  bitnet_cvt_kernel<<<(unsigned)(nIn / (256 * 4)), 256, 0, stream>>>(input, ain);

  // Phase 3: WMMA GEMM + fused scale/clamp epilogue.
  dim3 grid(N_DIM / BN, M_DIM / BM);
  bitnet_gemm_kernel<<<grid, 256, 0, stream>>>(ain, wb, cscale, out);
}
